// TemporalMamba_28793460752643
// MI455X (gfx1250) — matrix operand order repaired and checked by reading, not compiled, not run on hardware
//
#include <hip/hip_runtime.h>
#include <hip/hip_bf16.h>
#include <math.h>

typedef __attribute__((ext_vector_type(16))) _Float16 v16h;
typedef __attribute__((ext_vector_type(8)))  float    v8f;

#define BSZ     2048
#define TLEN    64
#define MT      (BSZ*TLEN)      /* 131072 rows */
#define DMODEL  256
#define DINNER  512
#define DSTATE  16

__device__ __forceinline__ float sigmoidf_(float x){ return 1.f/(1.f+__expf(-x)); }
__device__ __forceinline__ float siluf_(float x){ return x*sigmoidf_(x); }
__device__ __forceinline__ float softplusf_(float x){ return (x>20.f)? x : log1pf(__expf(x)); }

// ---------------- input projection (K=12) + LayerNorm(256) ----------------
__global__ void k_inproj_ln(const float* __restrict__ x, const float* __restrict__ w,
                            const float* __restrict__ b, const float* __restrict__ g,
                            const float* __restrict__ beta, float* __restrict__ out){
  int row = blockIdx.x; int j = threadIdx.x;
  __shared__ float xin[12];
  __shared__ float red[256];
  if (j < 12) xin[j] = x[(size_t)row*12 + j];
  __syncthreads();
  float acc = b[j];
  #pragma unroll
  for (int k=0;k<12;++k) acc += xin[k]*w[j*12+k];
  red[j]=acc; __syncthreads();
  for (int off=128; off>0; off>>=1){ if(j<off) red[j]+=red[j+off]; __syncthreads(); }
  float mean = red[0]*(1.f/256.f); __syncthreads();
  float d = acc-mean; red[j]=d*d; __syncthreads();
  for (int off=128; off>0; off>>=1){ if(j<off) red[j]+=red[j+off]; __syncthreads(); }
  float var = red[0]*(1.f/256.f);
  out[(size_t)row*DMODEL + j] = d*rsqrtf(var+1e-5f)*g[j]+beta[j];
}

// ---------------- LayerNorm(256) over rows of h ----------------
__global__ void k_ln256(const float* __restrict__ h, const float* __restrict__ g,
                        const float* __restrict__ beta, float* __restrict__ out){
  int row=blockIdx.x; int j=threadIdx.x;
  __shared__ float red[256];
  float v = h[(size_t)row*DMODEL+j];
  red[j]=v; __syncthreads();
  for (int off=128; off>0; off>>=1){ if(j<off) red[j]+=red[j+off]; __syncthreads(); }
  float mean=red[0]*(1.f/256.f); __syncthreads();
  float d=v-mean; red[j]=d*d; __syncthreads();
  for (int off=128; off>0; off>>=1){ if(j<off) red[j]+=red[j+off]; __syncthreads(); }
  float var=red[0]*(1.f/256.f);
  out[(size_t)row*DMODEL+j]=d*rsqrtf(var+1e-5f)*g[j]+beta[j];
}

// ---------------- WMMA GEMM: C[M,N] (+)= A[M,K] * W[N,K]^T ----------------
// K must be a multiple of 32; rows 16B-aligned. One 16x(16*NT) C strip per wave,
// 8 waves/block. A fragment shared across NT column tiles (4 v_wmma per K-step
// for NT=4). All fragment gathers are float4 (global_load_b128), f32->f16 in regs.
template<int NT>
__global__ void k_wmma_gemm(const float* __restrict__ A, const float* __restrict__ W,
                            float* __restrict__ C, int M, int N, int K, int addC){
  int wid  = threadIdx.x >> 5;
  int lane = threadIdx.x & 31;
  int ng   = N / (16*NT);
  int tile = blockIdx.x*8 + wid;
  int tm   = (tile / ng) << 4;
  int tn0  = (tile % ng) * (16*NT);
  const float* arow = A + (size_t)(tm + (lane&15))*K;
  int ha = (lane & 16) ? 8  : 0;   // A: half-wave K offset
  int hb = (lane & 16) ? 8  : 0;   // B: half-wave K offset
  v8f acc[NT];
  #pragma unroll
  for (int u=0;u<NT;++u) acc[u] = (v8f){};
  for (int kk=0; kk<K; kk+=32){
    // A fragment: slots 0..7 = K(base..base+7), slots 8..15 = K(base+16..base+23)
    const float4* ap = reinterpret_cast<const float4*>(arow + kk + ha);
    float4 a0 = ap[0], a1 = ap[1], a2 = ap[4], a3 = ap[5];
    v16h af;
    af[0]=(_Float16)a0.x; af[1]=(_Float16)a0.y; af[2]=(_Float16)a0.z; af[3]=(_Float16)a0.w;
    af[4]=(_Float16)a1.x; af[5]=(_Float16)a1.y; af[6]=(_Float16)a1.z; af[7]=(_Float16)a1.w;
    af[8]=(_Float16)a2.x; af[9]=(_Float16)a2.y; af[10]=(_Float16)a2.z; af[11]=(_Float16)a2.w;
    af[12]=(_Float16)a3.x; af[13]=(_Float16)a3.y; af[14]=(_Float16)a3.z; af[15]=(_Float16)a3.w;
    #pragma unroll
    for (int u=0;u<NT;++u){
      // B fragment: lane -> column, slots 0..7 = K(base..base+7), slots 8..15 = K(base+16..base+23)
      const float* wrow = W + (size_t)(tn0 + u*16 + (lane&15))*K;
      const float4* bp = reinterpret_cast<const float4*>(wrow + kk + hb);
      float4 b0 = bp[0], b1 = bp[1], b2 = bp[4], b3 = bp[5];
      v16h bf;
      bf[0]=(_Float16)b0.x; bf[1]=(_Float16)b0.y; bf[2]=(_Float16)b0.z; bf[3]=(_Float16)b0.w;
      bf[4]=(_Float16)b1.x; bf[5]=(_Float16)b1.y; bf[6]=(_Float16)b1.z; bf[7]=(_Float16)b1.w;
      bf[8]=(_Float16)b2.x; bf[9]=(_Float16)b2.y; bf[10]=(_Float16)b2.z; bf[11]=(_Float16)b2.w;
      bf[12]=(_Float16)b3.x; bf[13]=(_Float16)b3.y; bf[14]=(_Float16)b3.z; bf[15]=(_Float16)b3.w;
      acc[u] = __builtin_amdgcn_wmma_f32_16x16x32_f16(false, af, false, bf, (short)0,
                                                      acc[u], false, false);
    }
  }
  int orow = tm + ((lane&16)?8:0);
  #pragma unroll
  for (int u=0;u<NT;++u){
    int ocol = tn0 + u*16 + (lane&15);
    #pragma unroll
    for (int r=0;r<8;++r){
      size_t idx = (size_t)(orow + r)*N + ocol;
      if (addC) C[idx] += acc[u][r]; else C[idx] = acc[u][r];
    }
  }
}

// ---------------- dt projection: K=16 VALU kernel ----------------
// dt[row,n] = softplus(dt_b[n] + sum_k xdbl[row*48+k] * dt_w[n*16+k])
__global__ void k_dtproj(const float* __restrict__ xdbl, const float* __restrict__ dtw,
                         const float* __restrict__ dtb_bias, float* __restrict__ dtb){
  int row = blockIdx.x; int n = threadIdx.x;   // 512 threads
  __shared__ float xr[16];
  if (n < 16) xr[n] = xdbl[(size_t)row*48 + n];
  __syncthreads();
  float s = dtb_bias[n];
  const float4* wp = reinterpret_cast<const float4*>(dtw + n*16);
  #pragma unroll
  for (int q=0;q<4;++q){
    float4 w4 = wp[q];
    s += xr[q*4+0]*w4.x + xr[q*4+1]*w4.y + xr[q*4+2]*w4.z + xr[q*4+3]*w4.w;
  }
  dtb[(size_t)row*DINNER + n] = softplusf_(s);
}

// ---------------- depthwise causal conv1d(k=4) + bias + SiLU ----------------
__global__ void k_conv_silu(const float* __restrict__ xz, const float* __restrict__ cw,
                            const float* __restrict__ cb, float* __restrict__ xi){
  int idx = blockIdx.x*256 + threadIdx.x;        // b*32768 + t*512 + d
  int d = idx & 511;
  int t = (idx >> 9) & 63;
  int b = idx >> 15;
  float s = cb[d];
  #pragma unroll
  for (int j=0;j<4;++j){
    int tt = t - 3 + j;
    if (tt >= 0) s += cw[d*4+j] * xz[((size_t)(b*TLEN+tt))*1024 + d];
  }
  xi[(size_t)idx] = siluf_(s);
}

// ---------------- selective scan (sequential over T, parallel over B x D_INNER) -----
__global__ void k_scan(const float* __restrict__ dtb, const float* __restrict__ xdbl,
                       float* __restrict__ xi, const float* __restrict__ xz,
                       const float* __restrict__ A_log, const float* __restrict__ Dv){
  int b = blockIdx.x; int d = threadIdx.x;     // 512 threads
  __shared__ float BC[32];
  float A[DSTATE];
  #pragma unroll
  for (int n=0;n<DSTATE;++n) A[n] = -__expf(A_log[d*DSTATE+n]);
  float Dd = Dv[d];
  float h[DSTATE];
  #pragma unroll
  for (int n=0;n<DSTATE;++n) h[n]=0.f;
  for (int t=0;t<TLEN;++t){
    size_t row = (size_t)b*TLEN + t;
    __syncthreads();
    if (d < 32) BC[d] = xdbl[row*48 + 16 + d];   // B_t (16) then C_t (16)
    __syncthreads();
    float dtv = dtb[row*DINNER + d];
    float xv  = xi [row*DINNER + d];
    float zv  = xz [row*1024 + DINNER + d];
    float dx  = dtv*xv;
    float y = 0.f;
    #pragma unroll
    for (int n=0;n<DSTATE;++n){
      h[n] = h[n]*__expf(dtv*A[n]) + dx*BC[n];
      y += h[n]*BC[16+n];
    }
    y = (y + Dd*xv) * siluf_(zv);
    xi[row*DINNER + d] = y;
  }
}

// ---------------- output heads: one block (256 thr) per batch element ----------------
__global__ void k_heads(const float* __restrict__ h,
                        const float* qw1, const float* qb1, const float* qw2, const float* qb2,
                        const float* rw1, const float* rb1, const float* rw2, const float* rb2,
                        const float* rqs, const float* rrs, const float* rqo,
                        const float* kw,  const float* kb,
                        const float* ew,  const float* eb, const float* eg, const float* ebe,
                        const int* __restrict__ cls,
                        float* Qpos, float* Qsiz, float* Qori, float* Rpos, float* Rsiz,
                        float* Rori, float* Kap, float* Emb){
  int b = blockIdx.x; int j = threadIdx.x;
  __shared__ float hl[256];
  __shared__ float hid[64];
  __shared__ float raw[24];
  __shared__ float Lm[36];
  __shared__ float red[256];
  __shared__ float mv[2];
  hl[j] = h[((size_t)b*TLEN + (TLEN-1))*DMODEL + j];
  __syncthreads();
  // ---- Q_pos head (6x6 cholesky) ----
  if (j < 64){
    float s = qb1[j];
    for (int k=0;k<256;++k) s += hl[k]*qw1[j*256+k];
    hid[j] = siluf_(s);
  }
  __syncthreads();
  if (j < 21){
    float s = qb2[j];
    for (int k=0;k<64;++k) s += hid[k]*qw2[j*64+k];
    raw[j]=s;
  }
  __syncthreads();
  if (j < 36) Lm[j]=0.f;
  __syncthreads();
  if (j < 21){
    int r=0; while ((r+1)*(r+2)/2 <= j) ++r;
    int c = j - r*(r+1)/2;
    float v = raw[j];
    if (r==c) v = softplusf_(v) + 0.1f + 1e-5f;
    Lm[r*6+c]=v;
  }
  __syncthreads();
  if (j < 36){
    int r=j/6, c=j%6;
    int kk = (r<c? r:c);
    float s=0.f;
    for (int k=0;k<=kk;++k) s += Lm[r*6+k]*Lm[c*6+k];
    Qpos[(size_t)b*36+j]=s;
  }
  __syncthreads();
  // ---- R_pos head (3x3 cholesky) ----
  if (j < 64){
    float s = rb1[j];
    for (int k=0;k<256;++k) s += hl[k]*rw1[j*256+k];
    hid[j] = siluf_(s);
  }
  __syncthreads();
  if (j < 6){
    float s = rb2[j];
    for (int k=0;k<64;++k) s += hid[k]*rw2[j*64+k];
    raw[j]=s;
  }
  __syncthreads();
  if (j < 9) Lm[j]=0.f;
  __syncthreads();
  if (j < 6){
    int r=0; while ((r+1)*(r+2)/2 <= j) ++r;
    int c = j - r*(r+1)/2;
    float v = raw[j];
    if (r==c) v = softplusf_(v) + 0.1f + 1e-5f;
    Lm[r*3+c]=v;
  }
  __syncthreads();
  if (j < 9){
    int r=j/3, c=j%3;
    int kk=(r<c? r:c);
    float s=0.f;
    for (int k=0;k<=kk;++k) s += Lm[r*3+k]*Lm[c*3+k];
    Rpos[(size_t)b*9+j]=s;
  }
  // ---- class-indexed diagonal sizes + orientation ----
  int cid = cls[b];
  if (j < 9){
    int r=j/3, c=j%3;
    Qsiz[(size_t)b*9+j] = (r==c)? __expf(rqs[cid*3+r]) : 0.f;
    Rsiz[(size_t)b*9+j] = (r==c)? __expf(rrs[cid*3+r]) : 0.f;
  }
  if (j < 4){
    int r=j/2, c=j%2;
    Qori[(size_t)b*4+j] = (r==c)? __expf(rqo[0]) : 0.f;
  }
  // ---- kappa / R_ori ----
  __syncthreads();
  red[j] = hl[j]*kw[j];
  __syncthreads();
  for (int off=128; off>0; off>>=1){ if(j<off) red[j]+=red[j+off]; __syncthreads(); }
  if (j==0){
    float kp = softplusf_(red[0]+kb[0]) + 0.001f;
    Kap[b]=kp; Rori[b]=1.f/kp;
  }
  __syncthreads();
  // ---- embedding + LN(64) ----
  if (j<64){
    float s=eb[j];
    for (int k=0;k<256;++k) s += hl[k]*ew[j*256+k];
    hid[j]=s;
  }
  __syncthreads();
  if (j==0){
    float s=0.f; for(int k=0;k<64;++k) s+=hid[k];
    float m=s*(1.f/64.f); float v=0.f;
    for(int k=0;k<64;++k){ float dd=hid[k]-m; v+=dd*dd; }
    mv[0]=m; mv[1]=v*(1.f/64.f);
  }
  __syncthreads();
  if (j<64) Emb[(size_t)b*64+j] = (hid[j]-mv[0])*rsqrtf(mv[1]+1e-5f)*eg[j]+ebe[j];
}

extern "C" void kernel_launch(void* const* d_in, const int* in_sizes, int n_in,
                              void* d_out, int out_size, void* d_ws, size_t ws_size,
                              hipStream_t stream){
  (void)in_sizes; (void)n_in; (void)out_size; (void)ws_size;
  // input order: track_history, params (insertion order, layers nested), class_ids
  const float* track = (const float*)d_in[0];
  float* ws = (float*)d_ws;
  float* h    = ws;                      // MT*256
  float* xln  = ws + 33554432ull;        // MT*256
  float* xz   = ws + 67108864ull;        // MT*1024
  float* xi   = ws + 201326592ull;       // MT*512   (y written in-place)
  float* xdbl = ws + 268435456ull;       // MT*48
  float* dtb  = ws + 274726912ull;       // MT*512

  k_inproj_ln<<<MT,256,0,stream>>>(track, (const float*)d_in[1], (const float*)d_in[2],
                                   (const float*)d_in[3], (const float*)d_in[4], h);
  for (int L=0; L<3; ++L){
    int li = 22 + L*11;
    const float* ln_g   = (const float*)d_in[li+0];
    const float* ln_b   = (const float*)d_in[li+1];
    const float* in_w   = (const float*)d_in[li+2];
    const float* conv_w = (const float*)d_in[li+3];
    const float* conv_b = (const float*)d_in[li+4];
    const float* xprojw = (const float*)d_in[li+5];
    const float* dt_w   = (const float*)d_in[li+6];
    const float* dt_b   = (const float*)d_in[li+7];
    const float* A_log  = (const float*)d_in[li+8];
    const float* Dvec   = (const float*)d_in[li+9];
    const float* out_w  = (const float*)d_in[li+10];

    k_ln256<<<MT,256,0,stream>>>(h, ln_g, ln_b, xln);
    // xz = xln @ in_w.T : M=131072, N=1024, K=256 -> 8192 * 16 strip-tiles / 8 waves
    k_wmma_gemm<4><<<(8192*16)/8,256,0,stream>>>(xln, in_w, xz, MT,1024,256, 0);
    k_conv_silu<<<(MT*DINNER)/256,256,0,stream>>>(xz, conv_w, conv_b, xi);
    // xdbl = xi @ xproj.T : N=48 (3 tiles), K=512
    k_wmma_gemm<1><<<(8192*3)/8,256,0,stream>>>(xi, xprojw, xdbl, MT,48,512, 0);
    // dt = softplus(xdbl[:, :16] @ dt_w.T + dt_b) : K=16 -> VALU kernel
    k_dtproj<<<MT,DINNER,0,stream>>>(xdbl, dt_w, dt_b, dtb);
    k_scan<<<BSZ,DINNER,0,stream>>>(dtb, xdbl, xi, xz, A_log, Dvec);
    // h += y @ out_w.T : N=256, K=512 -> 8192 * 4 strip-tiles / 8 waves
    k_wmma_gemm<4><<<(8192*4)/8,256,0,stream>>>(xi, out_w, h, MT,256,512, 1);
  }
  float* out = (float*)d_out;
  k_heads<<<BSZ,256,0,stream>>>(h,
      (const float*)d_in[5],  (const float*)d_in[6],  (const float*)d_in[7],  (const float*)d_in[8],
      (const float*)d_in[9],  (const float*)d_in[10], (const float*)d_in[11], (const float*)d_in[12],
      (const float*)d_in[13], (const float*)d_in[14], (const float*)d_in[15],
      (const float*)d_in[16], (const float*)d_in[17],
      (const float*)d_in[18], (const float*)d_in[19], (const float*)d_in[20], (const float*)d_in[21],
      (const int*)d_in[55],
      out, out+73728, out+92160, out+100352, out+118784, out+137216, out+139264, out+141312);
}
